// GATLayer_82463372083609
// MI455X (gfx1250) — compile-verified, matrix operand order checked by reading, not run on hardware
//
#include <hip/hip_runtime.h>
#include <hip/hip_bf16.h>
#include <cstdint>

#define NN   4096
#define DD   512
#define HHN  8
#define NHID 64
#define EE   131072

typedef __attribute__((ext_vector_type(16))) __bf16 v16bf;
typedef __attribute__((ext_vector_type(8)))  __bf16 v8bf;
typedef __attribute__((ext_vector_type(8)))  float  v8f;
typedef __attribute__((ext_vector_type(8)))  unsigned short v8us;
typedef __attribute__((ext_vector_type(4)))  unsigned int v4u;
typedef __attribute__((ext_vector_type(8)))  unsigned int v8u;

static __device__ __forceinline__ unsigned short f2bf(float f) {
  unsigned u = __float_as_uint(f);
  u += 0x7FFFu + ((u >> 16) & 1u);          // round-to-nearest-even
  return (unsigned short)(u >> 16);
}
static __device__ __forceinline__ float bf2f(unsigned short s) {
  return __uint_as_float(((unsigned)s) << 16);
}

// Build a 16-element bf16 WMMA fragment from two contiguous 8-element (16B)
// chunks at p[0..7] and p[16..23] (16-bit A/B lane layout: K=kb+0..7 in
// regs 0..3, K=kb+16..23 in regs 4..7).
static __device__ __forceinline__ v16bf load_frag(const unsigned short* p) {
  v8us lo = *(const v8us*)(p);
  v8us hi = *(const v8us*)(p + 16);
  v8bf lb = __builtin_bit_cast(v8bf, lo);
  v8bf hb = __builtin_bit_cast(v8bf, hi);
  v16bf r;
#pragma unroll
  for (int i = 0; i < 8; ++i) { r[i] = lb[i]; r[i + 8] = hb[i]; }
  return r;
}

static __device__ __forceinline__ v8f wmma_bf16(v16bf a, v16bf b, v8f c) {
  return __builtin_amdgcn_wmma_f32_16x16x32_bf16(
      false, a, false, b, (short)0, c, false, false);
}

// LDS byte offset of a __shared__ object: generic LDS addresses carry the
// LDS offset in bits [31:0] (aperture bits live in [63:32]).
static __device__ __forceinline__ unsigned lds_off(const void* p) {
  return (unsigned)(unsigned long long)p;
}

// Issue a 2D TDM transfer Global->LDS. All dims/strides in elements
// (data_size = 2 bytes). Descriptor layout per CDNA5 ISA 8.3/8.4.
static __device__ __forceinline__ void tdm_load_2d(
    unsigned lds_addr, const void* gptr,
    unsigned tile_d0, unsigned tile_d1,
    unsigned tensor_d0, unsigned tensor_d1,
    unsigned long long stride0_elems) {
  unsigned long long ga = (unsigned long long)gptr;
  v4u g0;
  g0[0] = 1u;                                        // count=1, user mode
  g0[1] = lds_addr;                                  // lds_addr
  g0[2] = (unsigned)(ga & 0xFFFFFFFFu);              // global_addr[31:0]
  g0[3] = (unsigned)((ga >> 32) & 0x01FFFFFFu)       // global_addr[56:32]
          | (2u << 30);                              // type = 2 ("image")
  v8u g1;
  g1[0] = (1u << 16);                                // data_size=1 (2 bytes)
  g1[1] = (tensor_d0 & 0xFFFFu) << 16;               // tensor_dim0[15:0]
  g1[2] = ((tensor_d0 >> 16) & 0xFFFFu)              // tensor_dim0[31:16]
          | ((tensor_d1 & 0xFFFFu) << 16);           // tensor_dim1[15:0]
  g1[3] = ((tensor_d1 >> 16) & 0xFFFFu)              // tensor_dim1[31:16]
          | ((tile_d0 & 0xFFFFu) << 16);             // tile_dim0
  g1[4] = (tile_d1 & 0xFFFFu);                       // tile_dim1 (tile_dim2=0)
  g1[5] = (unsigned)(stride0_elems & 0xFFFFFFFFu);   // tensor_dim0_stride lo
  g1[6] = (unsigned)((stride0_elems >> 32) & 0xFFFFu); // stride hi, dim1_stride=0
  g1[7] = 0u;
  asm volatile("tensor_load_to_lds %0, %1" :: "s"(g0), "s"(g1) : "memory");
}

// ---------------------------------------------------------------------------
// Conversion / preprocessing kernels
// ---------------------------------------------------------------------------
__global__ void cvt_h_kernel(const float* __restrict__ h,
                             unsigned short* __restrict__ hbf, int n) {
  int i = blockIdx.x * blockDim.x + threadIdx.x;
  if (i < n) hbf[i] = f2bf(h[i]);
}

// WT layout: [mat(3)][head][j(64)][d(512)]
__global__ void cvt_wqkv_kernel(const float* __restrict__ Wq,
                                const float* __restrict__ Wk,
                                const float* __restrict__ Wv,
                                unsigned short* __restrict__ WT) {
  int i = blockIdx.x * blockDim.x + threadIdx.x;
  const int per = HHN * NHID * DD;
  if (i >= 3 * per) return;
  int mat  = i / per;
  int rem  = i - mat * per;
  int head = rem / (NHID * DD);
  int r2   = rem - head * (NHID * DD);
  int j    = r2 / DD;
  int d    = r2 - j * DD;
  const float* W = (mat == 0) ? Wq : (mat == 1) ? Wk : Wv;
  WT[i] = f2bf(W[((size_t)head * DD + d) * NHID + j]);
}

__global__ void cvt_wo_kernel(const float* __restrict__ Wo,
                              unsigned short* __restrict__ WoT) {
  int i = blockIdx.x * blockDim.x + threadIdx.x;
  if (i >= DD * DD) return;
  int j = i >> 9, d = i & 511;              // WoT[j][d] = Wo[d][j]
  WoT[i] = f2bf(Wo[(size_t)d * DD + j]);
}

__global__ void adjb_init_kernel(const float* __restrict__ adj,
                                 unsigned short* __restrict__ adjb, int n) {
  int i = blockIdx.x * blockDim.x + threadIdx.x;
  if (i < n) adjb[i] = (adj[i] > 0.0f) ? (unsigned short)0u
                                       : (unsigned short)0xFF80u;
}

__global__ void adjb_scatter_kernel(const int* __restrict__ er,
                                    const int* __restrict__ ec,
                                    const int* __restrict__ et,
                                    const float* __restrict__ emb,
                                    unsigned short* __restrict__ adjb) {
  int e = blockIdx.x * blockDim.x + threadIdx.x;
  if (e < EE) adjb[(size_t)er[e] * NN + ec[e]] = f2bf(emb[et[e]]);
}

// ---------------------------------------------------------------------------
// Q/K/V projection GEMM: one wave computes one 16x16 tile, K=512 in 16 WMMAs
// ---------------------------------------------------------------------------
__global__ __launch_bounds__(32) void proj_kernel(
    const unsigned short* __restrict__ hbf, const unsigned short* __restrict__ WT,
    const float* __restrict__ bq, const float* __restrict__ bk,
    unsigned short* __restrict__ Qbf, unsigned short* __restrict__ Kbf,
    unsigned short* __restrict__ VT) {
  const int lane = threadIdx.x;
  const int nl   = lane & 15;
  const int kb   = (lane & 16) ? 8 : 0;
  const int rofs = (lane & 16) ? 8 : 0;
  const int r0   = blockIdx.x * 16;
  const int c0   = blockIdx.y * 16;
  const int head = blockIdx.z / 3;
  const int mat  = blockIdx.z % 3;

  const unsigned short* W = WT + ((size_t)mat * HHN + head) * (NHID * DD)
                               + (size_t)(c0 + nl) * DD;
  const unsigned short* A = hbf + (size_t)(r0 + nl) * DD;

  v8f c = {};
#pragma unroll 4
  for (int k0 = 0; k0 < DD; k0 += 32) {
    v16bf a = load_frag(A + k0 + kb);
    v16bf b = load_frag(W + k0 + kb);
    c = wmma_bf16(a, b, c);
  }

  if (mat == 2) {
#pragma unroll
    for (int e = 0; e < 8; ++e)
      VT[((size_t)head * NHID + c0 + nl) * NN + (r0 + rofs + e)] = f2bf(c[e]);
  } else {
    const float* bias = (mat == 0) ? bq : bk;
    float bv = bias[head * NHID + c0 + nl];
    unsigned short* dst = (mat == 0) ? Qbf : Kbf;
#pragma unroll
    for (int e = 0; e < 8; ++e)
      dst[((size_t)head * NN + r0 + rofs + e) * NHID + (c0 + nl)] =
          f2bf(c[e] + bv);
  }
}

// ---------------------------------------------------------------------------
// Fused masked attention, flash-style. Workgroup = 4 waves = 64 query rows of
// one head. K/V 32-key tiles are staged into double-buffered LDS by the
// Tensor Data Mover (issued by wave 0, TENSORcnt-synchronized), shared by all
// 4 waves. Each wave: QK^T WMMA from LDS, adjb mask+bias, online softmax,
// P transposed through LDS, P.V WMMA from LDS.
// ---------------------------------------------------------------------------
__global__ __launch_bounds__(128) void attn_kernel(
    const unsigned short* __restrict__ Qbf, const unsigned short* __restrict__ Kbf,
    const unsigned short* __restrict__ VT,  const unsigned short* __restrict__ adjb,
    unsigned short* __restrict__ attnbf) {
  __shared__ __align__(16) unsigned short Klds[2][32 * NHID];  // 2 x 4KB
  __shared__ __align__(16) unsigned short Vlds[2][NHID * 32];  // 2 x 4KB
  __shared__ __align__(16) unsigned short Plds[4][16 * 32];    // 4 x 1KB

  const int tid  = threadIdx.x;
  const int wid  = tid >> 5;
  const int lane = tid & 31;
  const int nl   = lane & 15;
  const int kb   = (lane & 16) ? 8 : 0;
  const int rofs = (lane & 16) ? 8 : 0;
  const int rblk = blockIdx.x * 64 + wid * 16;   // this wave's 16 query rows
  const int head = blockIdx.y;

  const unsigned short* Qb = Qbf + ((size_t)head * NN + rblk + nl) * NHID;
  const unsigned short* Kb = Kbf + (size_t)head * NN * NHID;
  const unsigned short* Vb = VT  + (size_t)head * NHID * NN;
  unsigned short* Pl = &Plds[wid][0];

  v16bf qa0 = load_frag(Qb + kb);        // K dims 0..31
  v16bf qa1 = load_frag(Qb + 32 + kb);   // K dims 32..63

  float m_i[8], l_i[8];
  v8f acc[4];
  v8f vz = {};
#pragma unroll
  for (int e = 0; e < 8; ++e) { m_i[e] = -3.0e38f; l_i[e] = 0.0f; }
#pragma unroll
  for (int t = 0; t < 4; ++t) acc[t] = vz;

  // prologue: stage first K/V tile into buffer 0
  if (wid == 0) {
    tdm_load_2d(lds_off(&Klds[0][0]), Kb, NHID, 32, NHID, NN, NHID);
    tdm_load_2d(lds_off(&Vlds[0][0]), Vb, 32, NHID, NN, NHID, NN);
    __builtin_amdgcn_s_wait_tensorcnt(0);
  }
  __syncthreads();

  for (int j0 = 0; j0 < NN; j0 += 32) {
    const int b = (j0 >> 5) & 1;
    // prefetch next tile into the other buffer (overlaps this iteration)
    if (wid == 0 && (j0 + 32) < NN) {
      tdm_load_2d(lds_off(&Klds[b ^ 1][0]), Kb + (size_t)(j0 + 32) * NHID,
                  NHID, 32, NHID, NN, NHID);
      tdm_load_2d(lds_off(&Vlds[b ^ 1][0]), Vb + (j0 + 32),
                  32, NHID, NN, NHID, NN);
    }

    // S = Q . K^T for keys j0..j0+31 (two 16x16 tiles) from LDS
    const unsigned short* K0 = &Klds[b][(size_t)nl * NHID];
    const unsigned short* K1 = K0 + 16 * NHID;
    v8f s0 = vz, s1 = vz;
    s0 = wmma_bf16(qa0, load_frag(K0 + kb), s0);
    s0 = wmma_bf16(qa1, load_frag(K0 + 32 + kb), s0);
    s1 = wmma_bf16(qa0, load_frag(K1 + kb), s1);
    s1 = wmma_bf16(qa1, load_frag(K1 + 32 + kb), s1);

    // mask + edge bias + row max (reduce over each 16-lane half)
    float p0[8], p1[8], rmax[8];
#pragma unroll
    for (int e = 0; e < 8; ++e) {
      size_t arow = (size_t)(rblk + rofs + e) * NN + j0;
      unsigned short a0 = adjb[arow + nl];
      unsigned short a1 = adjb[arow + 16 + nl];
      float v0 = (a0 == 0xFF80u) ? -1.125e15f
                                 : (s0[e] + bf2f(a0)) * (1.0f / 512.0f);
      float v1 = (a1 == 0xFF80u) ? -1.125e15f
                                 : (s1[e] + bf2f(a1)) * (1.0f / 512.0f);
      p0[e] = v0; p1[e] = v1;
      float mx = fmaxf(v0, v1);
      mx = fmaxf(mx, __shfl_xor(mx, 1));
      mx = fmaxf(mx, __shfl_xor(mx, 2));
      mx = fmaxf(mx, __shfl_xor(mx, 4));
      mx = fmaxf(mx, __shfl_xor(mx, 8));
      rmax[e] = mx;
    }
    float scl[8];
#pragma unroll
    for (int e = 0; e < 8; ++e) {
      float mn = fmaxf(m_i[e], rmax[e]);
      float sc = __expf(m_i[e] - mn);
      m_i[e] = mn; scl[e] = sc;
      float e0 = __expf(p0[e] - mn);
      float e1 = __expf(p1[e] - mn);
      p0[e] = e0; p1[e] = e1;
      float rs = e0 + e1;
      rs += __shfl_xor(rs, 1);
      rs += __shfl_xor(rs, 2);
      rs += __shfl_xor(rs, 4);
      rs += __shfl_xor(rs, 8);
      l_i[e] = l_i[e] * sc + rs;
    }
#pragma unroll
    for (int t = 0; t < 4; ++t)
#pragma unroll
      for (int e = 0; e < 8; ++e) acc[t][e] *= scl[e];

    // transpose P (C-layout -> A-layout) through this wave's LDS slice
#pragma unroll
    for (int e = 0; e < 8; ++e) {
      Pl[(rofs + e) * 32 + nl]      = f2bf(p0[e]);
      Pl[(rofs + e) * 32 + 16 + nl] = f2bf(p1[e]);
    }
    __syncthreads();
    v16bf pa = load_frag(Pl + nl * 32 + kb);
#pragma unroll
    for (int t = 0; t < 4; ++t) {
      const unsigned short* Vp = &Vlds[b][(size_t)(t * 16 + nl) * 32];
      acc[t] = wmma_bf16(pa, load_frag(Vp + kb), acc[t]);
    }

    // ensure prefetch landed; publish next buffer to all waves
    if (wid == 0) __builtin_amdgcn_s_wait_tensorcnt(0);
    __syncthreads();
  }

  // epilogue: divide by denominator, write concat-head layout [N][512]
#pragma unroll
  for (int t = 0; t < 4; ++t)
#pragma unroll
    for (int e = 0; e < 8; ++e) {
      float o = acc[t][e] / l_i[e];
      attnbf[(size_t)(rblk + rofs + e) * DD + head * NHID + t * 16 + nl] =
          f2bf(o);
    }
}

// ---------------------------------------------------------------------------
// Output projection: out = attn @ Wo + bo  (M=4096, K=512, N=512)
// ---------------------------------------------------------------------------
__global__ __launch_bounds__(32) void out_kernel(
    const unsigned short* __restrict__ attnbf,
    const unsigned short* __restrict__ WoT,
    const float* __restrict__ bo, float* __restrict__ out) {
  const int lane = threadIdx.x;
  const int nl   = lane & 15;
  const int kb   = (lane & 16) ? 8 : 0;
  const int rofs = (lane & 16) ? 8 : 0;
  const int r0   = blockIdx.x * 16;
  const int c0   = blockIdx.y * 16;

  const unsigned short* A = attnbf + (size_t)(r0 + nl) * DD;
  const unsigned short* B = WoT + (size_t)(c0 + nl) * DD;

  v8f c = {};
#pragma unroll 4
  for (int k0 = 0; k0 < DD; k0 += 32)
    c = wmma_bf16(load_frag(A + k0 + kb), load_frag(B + k0 + kb), c);

  float bv = bo[c0 + nl];
#pragma unroll
  for (int e = 0; e < 8; ++e)
    out[(size_t)(r0 + rofs + e) * DD + (c0 + nl)] = c[e] + bv;
}

// ---------------------------------------------------------------------------
extern "C" void kernel_launch(void* const* d_in, const int* in_sizes, int n_in,
                              void* d_out, int out_size, void* d_ws, size_t ws_size,
                              hipStream_t stream) {
  const float* h   = (const float*)d_in[0];
  const float* adj = (const float*)d_in[1];
  const int*   er  = (const int*)d_in[2];
  const int*   ec  = (const int*)d_in[3];
  const int*   et  = (const int*)d_in[4];
  const float* Wq  = (const float*)d_in[5];
  const float* bq  = (const float*)d_in[6];
  const float* Wk  = (const float*)d_in[7];
  const float* bk  = (const float*)d_in[8];
  const float* Wv  = (const float*)d_in[9];
  const float* emb = (const float*)d_in[10];
  const float* Wo  = (const float*)d_in[11];
  const float* bo  = (const float*)d_in[12];
  float* out = (float*)d_out;

  // workspace layout (bytes, 256B aligned), total ~56.6 MB
  char* ws = (char*)d_ws;
  unsigned short* adjb  = (unsigned short*)(ws);              // 33,554,432
  unsigned short* hbf   = (unsigned short*)(ws + 33554432);   //  4,194,304
  unsigned short* WT    = (unsigned short*)(ws + 37748736);   //  1,572,864
  unsigned short* WoT   = (unsigned short*)(ws + 39321600);   //    524,288
  unsigned short* Qbf   = (unsigned short*)(ws + 39845888);   //  4,194,304
  unsigned short* Kbf   = (unsigned short*)(ws + 44040192);   //  4,194,304
  unsigned short* VT    = (unsigned short*)(ws + 48234496);   //  4,194,304
  unsigned short* attnb = (unsigned short*)(ws + 52428800);   //  4,194,304

  cvt_h_kernel<<<(NN * DD) / 256, 256, 0, stream>>>(h, hbf, NN * DD);
  cvt_wqkv_kernel<<<(3 * HHN * NHID * DD) / 256, 256, 0, stream>>>(Wq, Wk, Wv, WT);
  cvt_wo_kernel<<<(DD * DD) / 256, 256, 0, stream>>>(Wo, WoT);
  adjb_init_kernel<<<(NN * NN) / 256, 256, 0, stream>>>(adj, adjb, NN * NN);
  adjb_scatter_kernel<<<EE / 256, 256, 0, stream>>>(er, ec, et, emb, adjb);

  proj_kernel<<<dim3(NN / 16, NHID / 16, 3 * HHN), 32, 0, stream>>>(
      hbf, WT, bq, bk, Qbf, Kbf, VT);
  attn_kernel<<<dim3(NN / 64, HHN), 128, 0, stream>>>(Qbf, Kbf, VT, adjb, attnb);
  out_kernel<<<dim3(NN / 16, DD / 16), 32, 0, stream>>>(attnb, WoT, bo, out);

  (void)in_sizes; (void)n_in; (void)out_size; (void)ws_size;
}
